// BASICEncoder_46007689675483
// MI455X (gfx1250) — matrix operand order repaired and checked by reading, not compiled, not run on hardware
//
#include <hip/hip_runtime.h>
#include <stdint.h>

#define B_   64
#define T_   512
#define D_   512
#define U_   512
#define G3   1536
#define NWG  32
#define TPB  128
#define SLICE_PITCH 1040                 // 512 bf16 = 1024B + 16B pad (bank-conflict-free b128)
#define SLICE_BYTES (48 * SLICE_PITCH)   // 3 gates x 16 units rows

typedef __attribute__((ext_vector_type(16))) __bf16 bf16x16;
typedef __attribute__((ext_vector_type(8)))  float  f32x8;

union FragU { uint4 q[2]; bf16x16 v; };

static __device__ inline unsigned short f2bf(float f) {
  unsigned u = __builtin_bit_cast(unsigned, f);
  u += 0x7FFFu + ((u >> 16) & 1u);       // round-to-nearest-even
  return (unsigned short)(u >> 16);
}
static __device__ inline float sigm(float x) { return 1.0f / (1.0f + __expf(-x)); }

// Grid-wide barrier: monotonic counter, counter zeroed by memsetAsync each launch.
static __device__ inline void gbar(unsigned* cnt, unsigned target) {
  __threadfence();
  __syncthreads();
  if (threadIdx.x == 0) {
    __hip_atomic_fetch_add(cnt, 1u, __ATOMIC_RELEASE, __HIP_MEMORY_SCOPE_AGENT);
    while (__hip_atomic_load(cnt, __ATOMIC_ACQUIRE, __HIP_MEMORY_SCOPE_AGENT) < target) {
      __builtin_amdgcn_s_sleep(1);
    }
  }
  __syncthreads();
}

// One 16x16 output tile vs three gate slices (rows 0-15 / 16-31 / 32-47 of the LDS
// weight slice), contracting K=512 in 16 WMMA k-steps. A from global (row-major,
// K-contiguous, row stride in elements), B from LDS.
static __device__ inline void mm3(const unsigned short* __restrict__ aRow,
                                  const char* __restrict__ bSlice,
                                  int colL, int kg,
                                  f32x8& a0, f32x8& a1, f32x8& a2) {
  const int r0 = (0  + colL) * SLICE_PITCH;
  const int r1 = (16 + colL) * SLICE_PITCH;
  const int r2 = (32 + colL) * SLICE_PITCH;
#pragma unroll
  for (int ks = 0; ks < 16; ++ks) {
    FragU fa;
    fa.q[0] = *(const uint4*)(aRow + ks * 32 + kg * 16);
    fa.q[1] = *(const uint4*)(aRow + ks * 32 + kg * 16 + 8);
    const int cb = ks * 64 + kg * 32;
    FragU fb0, fb1, fb2;
    fb0.q[0] = *(const uint4*)(bSlice + r0 + cb);
    fb0.q[1] = *(const uint4*)(bSlice + r0 + cb + 16);
    fb1.q[0] = *(const uint4*)(bSlice + r1 + cb);
    fb1.q[1] = *(const uint4*)(bSlice + r1 + cb + 16);
    fb2.q[0] = *(const uint4*)(bSlice + r2 + cb);
    fb2.q[1] = *(const uint4*)(bSlice + r2 + cb + 16);
    a0 = __builtin_amdgcn_wmma_f32_16x16x32_bf16(false, fa.v, false, fb0.v, (short)0, a0, false, false);
    a1 = __builtin_amdgcn_wmma_f32_16x16x32_bf16(false, fa.v, false, fb1.v, (short)0, a1, false, false);
    a2 = __builtin_amdgcn_wmma_f32_16x16x32_bf16(false, fa.v, false, fb2.v, (short)0, a2, false, false);
  }
}

__global__ void __launch_bounds__(TPB, 1)
gru_persistent(const unsigned short* __restrict__ xbf,
               const unsigned char*  __restrict__ mask,
               const unsigned short* __restrict__ Wt0,
               const unsigned short* __restrict__ Ut0,
               const unsigned short* __restrict__ Wt1,
               const unsigned short* __restrict__ Ut1,
               const float* __restrict__ b0,
               const float* __restrict__ b1,
               unsigned short* __restrict__ h0c0, unsigned short* __restrict__ h0c1,
               unsigned short* __restrict__ h1c0, unsigned short* __restrict__ h1c1,
               unsigned short* __restrict__ h0new,
               float* __restrict__ out,
               unsigned* __restrict__ cnt) {
  extern __shared__ char smem[];
  const int wg   = blockIdx.x;       // owns units [wg*16, wg*16+16)
  const int tid  = threadIdx.x;
  const int lane = tid & 31;
  const int wv   = tid >> 5;         // m-tile (16 batches)
  const int colL = lane & 15;
  const int kg   = lane >> 4;
  const int cu   = wg * 16 + colL;   // global unit column
  const int bA   = wv * 16 + colL;   // batch row for A-fragment loads
  const int bC0  = wv * 16 + kg * 8; // first batch row of this lane's C/D elems

  // ---- stage this WG's weight slices into LDS (resident for all 512 steps) ----
  const unsigned short* Wsrc[4] = {Wt0, Ut0, Wt1, Ut1};
  for (int id = tid; id < 4 * 48 * 64; id += TPB) {
    int s   = id / (48 * 64);
    int rem = id - s * (48 * 64);
    int gr  = rem >> 6;              // LDS row: gate*16 + unit
    int ck  = rem & 63;              // 16B chunk within K=512 row
    int n   = (gr >> 4) * U_ + wg * 16 + (gr & 15);
    *(uint4*)(smem + s * SLICE_BYTES + gr * SLICE_PITCH + ck * 16) =
        *(const uint4*)(Wsrc[s] + (size_t)n * D_ + ck * 8);
  }
  __syncthreads();

  // ---- per-lane gate biases (b[0] and b[1] fused where the math allows) ----
  const float bz0v = b0[cu]           + b0[G3 + cu];
  const float br0v = b0[U_ + cu]      + b0[G3 + U_ + cu];
  const float bci0 = b0[2 * U_ + cu];
  const float bch0 = b0[G3 + 2 * U_ + cu];
  const float bz1v = b1[cu]           + b1[G3 + cu];
  const float br1v = b1[U_ + cu]      + b1[G3 + U_ + cu];
  const float bci1 = b1[2 * U_ + cu];
  const float bch1 = b1[G3 + 2 * U_ + cu];

  float h0p[8], h1p[8], po[8];
#pragma unroll
  for (int i = 0; i < 8; ++i) { h0p[i] = 0.f; h1p[i] = 0.f; po[i] = 0.f; }

  for (int t = 0; t < T_; ++t) {
    const unsigned short* h0R = (t & 1) ? h0c1 : h0c0;
    unsigned short*       h0W = (t & 1) ? h0c0 : h0c1;
    const unsigned short* h1R = (t & 1) ? h1c1 : h1c0;
    unsigned short*       h1W = (t & 1) ? h1c0 : h1c1;

    int mreg[8];
#pragma unroll
    for (int i = 0; i < 8; ++i) mreg[i] = mask[(size_t)(bC0 + i) * T_ + t] != 0;

    // ================= layer 0 =================
    f32x8 az  = {0, 0, 0, 0, 0, 0, 0, 0};
    f32x8 ar  = {0, 0, 0, 0, 0, 0, 0, 0};
    f32x8 aci = {0, 0, 0, 0, 0, 0, 0, 0};
    f32x8 ach = {0, 0, 0, 0, 0, 0, 0, 0};
    mm3(xbf + (size_t)bA * (T_ * D_) + (size_t)t * D_, smem + 0 * SLICE_BYTES, colL, kg, az, ar, aci);
    mm3(h0R + bA * U_,                                 smem + 1 * SLICE_BYTES, colL, kg, az, ar, ach);
#pragma unroll
    for (int i = 0; i < 8; ++i) {
      const int bb = bC0 + i;
      float z    = sigm(az[i] + bz0v);
      float r    = sigm(ar[i] + br0v);
      float cand = tanhf(aci[i] + bci0 + r * (ach[i] + bch0));
      float hn   = z * h0p[i] + (1.0f - z) * cand;
      float hc   = mreg[i] ? hn : h0p[i];
      h0p[i] = hc;
      h0new[bb * U_ + cu] = f2bf(hn);   // layer-1 input uses un-carried h0n
      h0W[bb * U_ + cu]   = f2bf(hc);   // next step's recurrent state
    }
    gbar(cnt, (unsigned)(2 * t + 1) * NWG);

    // ================= layer 1 =================
    f32x8 bz  = {0, 0, 0, 0, 0, 0, 0, 0};
    f32x8 br_ = {0, 0, 0, 0, 0, 0, 0, 0};
    f32x8 bci = {0, 0, 0, 0, 0, 0, 0, 0};
    f32x8 bch = {0, 0, 0, 0, 0, 0, 0, 0};
    mm3(h0new + bA * U_, smem + 2 * SLICE_BYTES, colL, kg, bz, br_, bci);
    mm3(h1R + bA * U_,   smem + 3 * SLICE_BYTES, colL, kg, bz, br_, bch);
#pragma unroll
    for (int i = 0; i < 8; ++i) {
      const int bb = bC0 + i;
      float z    = sigm(bz[i] + bz1v);
      float r    = sigm(br_[i] + br1v);
      float cand = tanhf(bci[i] + bci1 + r * (bch[i] + bch1));
      float hn   = z * h1p[i] + (1.0f - z) * cand;
      float hc   = mreg[i] ? hn : h1p[i];
      float ov   = mreg[i] ? hn : po[i];
      h1p[i] = hc;
      po[i]  = ov;
      h1W[bb * U_ + cu] = f2bf(hc);
      out[(size_t)bb * ((size_t)T_ * U_) + (size_t)t * U_ + cu] = ov;
    }
    gbar(cnt, (unsigned)(2 * t + 2) * NWG);
  }

  // final states appended after output tensor
  float* h0f = out + (size_t)B_ * T_ * U_;
  float* h1f = h0f + (size_t)B_ * U_;
#pragma unroll
  for (int i = 0; i < 8; ++i) {
    const int bb = bC0 + i;
    h0f[bb * U_ + cu] = h0p[i];
    h1f[bb * U_ + cu] = h1p[i];
  }
}

// ---- one-time prep: f32 -> bf16 conversions ----
__global__ void cvt_x(const float* __restrict__ x, unsigned short* __restrict__ xb, int n) {
  int i = blockIdx.x * 256 + threadIdx.x;
  if (i < n) xb[i] = f2bf(x[i]);
}

// Transpose (D x 3U) f32 -> (3U x D) bf16 so K is contiguous per weight column.
__global__ void cvt_w(const float* __restrict__ W0, const float* __restrict__ U0,
                      const float* __restrict__ W1, const float* __restrict__ U1,
                      unsigned short* __restrict__ o0, unsigned short* __restrict__ o1,
                      unsigned short* __restrict__ o2, unsigned short* __restrict__ o3) {
  const float* src; unsigned short* dst;
  switch (blockIdx.y) {
    case 0:  src = W0; dst = o0; break;
    case 1:  src = U0; dst = o1; break;
    case 2:  src = W1; dst = o2; break;
    default: src = U1; dst = o3; break;
  }
  int i = blockIdx.x * 256 + threadIdx.x;
  if (i < G3 * D_) {
    int n = i / D_;
    int d = i - n * D_;
    dst[i] = f2bf(src[(size_t)d * G3 + n]);
  }
}

extern "C" void kernel_launch(void* const* d_in, const int* in_sizes, int n_in,
                              void* d_out, int out_size, void* d_ws, size_t ws_size,
                              hipStream_t stream) {
  (void)in_sizes; (void)n_in; (void)out_size; (void)ws_size;
  const float*         x    = (const float*)d_in[0];
  const unsigned char* mask = (const unsigned char*)d_in[1];
  const float*         W0   = (const float*)d_in[2];
  const float*         U0   = (const float*)d_in[3];
  const float*         b0   = (const float*)d_in[4];
  const float*         W1   = (const float*)d_in[5];
  const float*         U1   = (const float*)d_in[6];
  const float*         b1   = (const float*)d_in[7];
  float* out = (float*)d_out;

  char* ws = (char*)d_ws;
  size_t off = 0;
  auto alloc = [&](size_t bytes) -> char* {
    char* p = ws + off;
    off += (bytes + 255) & ~(size_t)255;
    return p;
  };
  unsigned short* xbf = (unsigned short*)alloc((size_t)B_ * T_ * D_ * 2);
  unsigned short* Wt0 = (unsigned short*)alloc((size_t)G3 * D_ * 2);
  unsigned short* Ut0 = (unsigned short*)alloc((size_t)G3 * D_ * 2);
  unsigned short* Wt1 = (unsigned short*)alloc((size_t)G3 * D_ * 2);
  unsigned short* Ut1 = (unsigned short*)alloc((size_t)G3 * D_ * 2);
  char* zbase = ws + off;                       // zero-init block: h0c0, h1c0, cnt
  unsigned short* h0c0 = (unsigned short*)alloc((size_t)B_ * U_ * 2);
  unsigned short* h1c0 = (unsigned short*)alloc((size_t)B_ * U_ * 2);
  unsigned*       cnt  = (unsigned*)alloc(256);
  size_t zbytes = (size_t)((ws + off) - zbase);
  unsigned short* h0c1 = (unsigned short*)alloc((size_t)B_ * U_ * 2);
  unsigned short* h1c1 = (unsigned short*)alloc((size_t)B_ * U_ * 2);
  unsigned short* h0n  = (unsigned short*)alloc((size_t)B_ * U_ * 2);

  hipMemsetAsync(zbase, 0, zbytes, stream);

  cvt_x<<<(B_ * T_ * D_ + 255) / 256, 256, 0, stream>>>(x, xbf, B_ * T_ * D_);
  dim3 gw((G3 * D_ + 255) / 256, 4);
  cvt_w<<<gw, 256, 0, stream>>>(W0, U0, W1, U1, Wt0, Ut0, Wt1, Ut1);

  (void)hipFuncSetAttribute((const void*)gru_persistent,
                            hipFuncAttributeMaxDynamicSharedMemorySize, 4 * SLICE_BYTES);
  gru_persistent<<<NWG, TPB, 4 * SLICE_BYTES, stream>>>(
      xbf, mask, Wt0, Ut0, Wt1, Ut1, b0, b1,
      h0c0, h0c1, h1c0, h1c1, h0n, out, cnt);
}
